// KMeansLSE_68049461838095
// MI455X (gfx1250) — matrix operand order repaired and checked by reading, not compiled, not run on hardware
//
#include <hip/hip_runtime.h>
#include <math.h>

// KMeans soft-min LSE: out[n] = ||x_n||^2 - logsumexp_k( 2*x_n.c_k - ||c_k||^2 )
// (== -logsumexp_k(-||x_n - c_k||^2), BETA = -1)
//
// v_wmma_f32_16x16x32_f16 GEMM core, fused online-logsumexp epilogue.
// Centroids pre-converted to f16 in d_ws (needs 528,384 bytes of workspace).
// B tiles staged global->LDS with async DMA (global_load_async_to_lds_b128,
// ASYNCcnt), double-buffered, consumed via ds_load_b128; two concurrent
// k-tile accumulator chains per wave for XDL ILP.

typedef __attribute__((ext_vector_type(16))) _Float16 v16h;
typedef __attribute__((ext_vector_type(8)))  _Float16 v8h;
typedef __attribute__((ext_vector_type(8)))  float    v8f;
typedef __attribute__((ext_vector_type(4)))  float    v4f;

#define DDIM 256
#define KCEN 1024
#define ROWS_PER_WAVE 16
#define WAVES_PER_BLOCK 8
#define ROWS_PER_BLOCK (ROWS_PER_WAVE * WAVES_PER_BLOCK)   // 128
#define KTILES (KCEN / 16)                                 // 64
#define LDS_ROW 264   // 256 f16 + 16B pad -> 528B row = 132 words = 4-bank stride/col

// ---------------------------------------------------------------------------
// Prep: centroids f32 -> f16 (row-major, K x D) + c2[k] = ||c_k||^2 (f32).
// ---------------------------------------------------------------------------
__global__ __launch_bounds__(256) void
KMeansLSE_prep_kernel(const float* __restrict__ cen,
                      _Float16* __restrict__ cws,
                      float* __restrict__ c2) {
    const int wave = threadIdx.x >> 5;
    const int lane = threadIdx.x & 31;
    const int k    = blockIdx.x * WAVES_PER_BLOCK + wave;   // grid = 128 blocks

    const float* cp = cen + (size_t)k * DDIM + lane * 8;
    v4f q0 = *(const v4f*)(cp);
    v4f q1 = *(const v4f*)(cp + 4);

    v8h h;
    float ss = 0.f;
#pragma unroll
    for (int j = 0; j < 4; ++j) {
        h[j]     = (_Float16)q0[j];
        h[4 + j] = (_Float16)q1[j];
        ss += q0[j] * q0[j] + q1[j] * q1[j];
    }
    *(v8h*)(cws + (size_t)k * DDIM + lane * 8) = h;

#pragma unroll
    for (int mask = 16; mask >= 1; mask >>= 1)
        ss += __shfl_xor(ss, mask, 32);
    if (lane == 0) c2[k] = ss;
}

__device__ __forceinline__ void lse_update(float (&mA)[8], float (&sA)[8],
                                           const v8f& acc, float c2v) {
#pragma unroll
    for (int v = 0; v < 8; ++v) {
        float t  = 2.0f * acc[v] - c2v;
        float mo = mA[v];
        float mn = fmaxf(mo, t);
        sA[v] = sA[v] * __expf(mo - mn) + __expf(t - mn);
        mA[v] = mn;
    }
}

// ---------------------------------------------------------------------------
// Main fused GEMM + logsumexp kernel.
// ---------------------------------------------------------------------------
__global__ __launch_bounds__(256) void
KMeansLSE_main_kernel(const float* __restrict__ x,
                      const _Float16* __restrict__ cws,
                      const float* __restrict__ c2,
                      float* __restrict__ out) {
    // [buffer][tile-in-pair][column(centroid)][d + pad]
    __shared__ _Float16 smem[2][2][16][LDS_ROW];

    const int wave = threadIdx.x >> 5;
    const int lane = threadIdx.x & 31;
    const int half = lane >> 4;       // 0: lanes 0-15, 1: lanes 16-31
    const int col  = lane & 15;

    const int m0   = blockIdx.x * ROWS_PER_BLOCK + wave * ROWS_PER_WAVE;
    const int rowg = m0 + col;        // this lane's A row (shared with lane^16)

    // Async-DMA a pair of 16-centroid B tiles (ktBase, ktBase+1) into smem[buf].
    // 2 tiles = 1024 x 16B units; 4 units per thread. Tracked by ASYNCcnt.
    auto stage_pair = [&](int ktBase, int buf) {
#pragma unroll
        for (int it = 0; it < 4; ++it) {
            const int u   = (int)threadIdx.x + it * 256;
            const int t   = u >> 9;          // tile within pair
            const int row = (u >> 5) & 15;   // centroid within tile
            const int un  = u & 31;          // 16B unit within row
            const _Float16* g =
                cws + (size_t)(((ktBase + t) * 16 + row)) * DDIM + un * 8;
            unsigned  ldsoff = (unsigned)(uintptr_t)(&smem[buf][t][row][un * 8]);
            unsigned long long ga = (unsigned long long)(uintptr_t)g;
            asm volatile("global_load_async_to_lds_b128 %0, %1, off"
                         :: "v"(ldsoff), "v"(ga) : "memory");
        }
    };

    // ---- Load A tile (16 rows x 256) once: f32 -> f16 fragments + ||x||^2.
    // 16-bit A 16x32 layout: lane holds row (lane&15);
    // elem e in [0,16) -> K = 16*half + e (matches contiguous B)
    v16h a[8];
    float x2p = 0.f;
    const float* xr = x + (size_t)rowg * DDIM;
#pragma unroll
    for (int f = 0; f < 8; ++f) {
        const float* p = xr + f * 32 + half * 16;
        v4f q0 = *(const v4f*)(p);
        v4f q1 = *(const v4f*)(p + 4);
        v4f q2 = *(const v4f*)(p + 8);
        v4f q3 = *(const v4f*)(p + 12);
#pragma unroll
        for (int j = 0; j < 4; ++j) {
            a[f][j]      = (_Float16)q0[j];
            a[f][4 + j]  = (_Float16)q1[j];
            a[f][8 + j]  = (_Float16)q2[j];
            a[f][12 + j] = (_Float16)q3[j];
            x2p += q0[j] * q0[j] + q1[j] * q1[j] + q2[j] * q2[j] + q3[j] * q3[j];
        }
    }
    const float x2full = x2p + __shfl_xor(x2p, 16, 32);

    float mA0[8], sA0[8], mA1[8], sA1[8];
#pragma unroll
    for (int v = 0; v < 8; ++v) {
        mA0[v] = -INFINITY; sA0[v] = 0.f;
        mA1[v] = -INFINITY; sA1[v] = 0.f;
    }

    // ---- Software pipeline: prefetch pair 0, then overlap DMA with WMMA.
    stage_pair(0, 0);

    for (int kt = 0; kt < KTILES; kt += 2) {
        const int buf = (kt >> 1) & 1;

        asm volatile("s_wait_asynccnt 0x0" ::: "memory");  // my DMA done
        __syncthreads();                                   // everyone's DMA done

        if (kt + 2 < KTILES) stage_pair(kt + 2, buf ^ 1);  // overlap next pair

        // B 32x16 f16 layout: lane = column N; lanes 0-15 K 0-15,
        // lanes 16-31 K 16-31 -> 32 contiguous bytes/lane/fragment from LDS.
        const _Float16* b0p = &smem[buf][0][col][half * 16];
        const _Float16* b1p = &smem[buf][1][col][half * 16];

        v8f acc0 = {0.f, 0.f, 0.f, 0.f, 0.f, 0.f, 0.f, 0.f};
        v8f acc1 = {0.f, 0.f, 0.f, 0.f, 0.f, 0.f, 0.f, 0.f};
#pragma unroll
        for (int f = 0; f < 8; ++f) {
            v8h l0 = *(const v8h*)(b0p + f * 32);
            v8h h0 = *(const v8h*)(b0p + f * 32 + 8);
            v8h l1 = *(const v8h*)(b1p + f * 32);
            v8h h1 = *(const v8h*)(b1p + f * 32 + 8);
            v16h b0 = __builtin_shufflevector(l0, h0, 0,1,2,3,4,5,6,7,8,9,10,11,12,13,14,15);
            v16h b1 = __builtin_shufflevector(l1, h1, 0,1,2,3,4,5,6,7,8,9,10,11,12,13,14,15);
            acc0 = __builtin_amdgcn_wmma_f32_16x16x32_f16(
                false, a[f], false, b0, (short)0, acc0, false, false);
            acc1 = __builtin_amdgcn_wmma_f32_16x16x32_f16(
                false, a[f], false, b1, (short)0, acc1, false, false);
        }

        const float c20 = c2[kt * 16 + col];
        const float c21 = c2[kt * 16 + 16 + col];
        lse_update(mA0, sA0, acc0, c20);   // t = 2*dot - c2 (x2 folded out)
        lse_update(mA1, sA1, acc1, c21);
    }

    // ---- Merge the two k-tile streams, then butterfly across the 16 lanes.
#pragma unroll
    for (int v = 0; v < 8; ++v) {
        float mn = fmaxf(mA0[v], mA1[v]);
        sA0[v] = sA0[v] * __expf(mA0[v] - mn) + sA1[v] * __expf(mA1[v] - mn);
        mA0[v] = mn;
    }
#pragma unroll
    for (int mask = 1; mask <= 8; mask <<= 1) {
#pragma unroll
        for (int v = 0; v < 8; ++v) {
            float mo = mA0[v], so = sA0[v];
            float m2 = __shfl_xor(mo, mask, 32);
            float s2 = __shfl_xor(so, mask, 32);
            float mn = fmaxf(mo, m2);
            sA0[v] = so * __expf(mo - mn) + s2 * __expf(m2 - mn);
            mA0[v] = mn;
        }
    }

    // ---- Write: lanes 0-7 emit rows 0-7 (v=lane), lanes 16-23 rows 8-15.
    const int vsel = lane & 7;
    const int row  = (lane & 7) + 8 * half;
    const float x2w = __shfl(x2full, (lane & 7) + 8 * half, 32);

    float mSel = mA0[0], sSel = sA0[0];
#pragma unroll
    for (int v = 1; v < 8; ++v)
        if (v == vsel) { mSel = mA0[v]; sSel = sA0[v]; }

    if ((lane & 15) < 8)
        out[m0 + row] = x2w - mSel - __logf(sSel);
}

extern "C" void kernel_launch(void* const* d_in, const int* in_sizes, int n_in,
                              void* d_out, int out_size, void* d_ws, size_t ws_size,
                              hipStream_t stream) {
    const float* x   = (const float*)d_in[0];       // (N, 256) f32
    const float* cen = (const float*)d_in[1];       // (1024, 256) f32
    float* out = (float*)d_out;                     // (N,) f32

    _Float16* cws = (_Float16*)d_ws;                             // 512 KB
    float*    c2  = (float*)((char*)d_ws + (size_t)KCEN * DDIM * sizeof(_Float16));

    const int N = in_sizes[0] / DDIM;               // 65536

    KMeansLSE_prep_kernel<<<KCEN / WAVES_PER_BLOCK, 256, 0, stream>>>(cen, cws, c2);
    KMeansLSE_main_kernel<<<N / ROWS_PER_BLOCK, 256, 0, stream>>>(x, cws, c2, out);
}
